// GNModel_13391708029669
// MI455X (gfx1250) — matrix operand order repaired and checked by reading, not compiled, not run on hardware
//
#include <hip/hip_runtime.h>
#include <hip/hip_bf16.h>

// ---------------------------------------------------------------------------
// GNN (MeshGraphNet-style) for MI455X / gfx1250, wave32 + WMMA bf16.
//
// Pipeline per kernel_launch:
//   0) pack all weights f32 -> bf16 in WMMA B-fragment lane order
//   1) pad+convert vertex_attr/edge_attr to bf16 (K padded to 32)
//   2) encoders: fused MLP(K->128->128)+LN  -> v(f32+bf16), e(f32+bf16)
//   3) 10x { edge fused MLP(384)+LN+residual (gathers v[src],v[dst]),
//            zero agg, scatter-add f32 atomics, agg->bf16,
//            vertex fused MLP(256)+LN+residual }
//   4) decoder fused MLP(128->128->12), no LN, f32 out
// ---------------------------------------------------------------------------

#define GN_N  50000
#define GN_E  800000
#define GN_H  128
#define GN_L  10
#define GN_VD 12
#define GN_ED 4
#define GN_OD 12

typedef __attribute__((ext_vector_type(16))) __bf16 bf16x16;
typedef __attribute__((ext_vector_type(8)))  __bf16 bf16x8;
typedef __attribute__((ext_vector_type(8)))  float  f32x8;

// ------------------------- small utility kernels ---------------------------

// Pack a row-major f32 weight [K x N] (logical Kpad x Npad, zero padded) into
// bf16 B-fragment order: element for (k = c*32+kk, n = nt*16+nn) goes to
//   P[(((c*ntiles + nt)*32 + ((kk/8)%2)*16 + nn)*16) + (kk%8) + (kk/16)*8]
// so each WMMA lane reads 16 contiguous bf16 (32B) per K-chunk.
__global__ __launch_bounds__(256) void pack_b_kernel(
    const float* __restrict__ W, __bf16* __restrict__ P,
    int K, int Kpad, int N, int Npad)
{
    int idx = blockIdx.x * 256 + threadIdx.x;
    int total = Kpad * Npad;
    if (idx >= total) return;
    int k = idx / Npad;
    int n = idx - k * Npad;
    float val = (k < K && n < N) ? W[(size_t)k * N + n] : 0.0f;
    int c = k >> 5, kk = k & 31;
    int nt = n >> 4, nn = n & 15;
    int ntiles = Npad >> 4;
    size_t p = ((size_t)((c * ntiles + nt) * 32 + ((kk >> 3) & 1) * 16 + nn) << 4) + (kk & 7) + ((kk >> 4) << 3);
    P[p] = (__bf16)val;
}

// Pad-convert f32 [rows x K] -> bf16 [rows x Kpad] (zero padded).
__global__ __launch_bounds__(256) void pad_rows_kernel(
    const float* __restrict__ in, __bf16* __restrict__ out,
    int rows, int K, int Kpad)
{
    long long idx = (long long)blockIdx.x * 256 + threadIdx.x;
    long long total = (long long)rows * Kpad;
    if (idx >= total) return;
    int r = (int)(idx / Kpad);
    int k = (int)(idx - (long long)r * Kpad);
    out[idx] = (k < K) ? (__bf16)in[(size_t)r * K + k] : (__bf16)0.0f;
}

__global__ __launch_bounds__(256) void zero_f32_kernel(float* __restrict__ p, long long n4)
{
    long long i = (long long)blockIdx.x * 256 + threadIdx.x;
    if (i >= n4) return;
    ((float4*)p)[i] = make_float4(0.f, 0.f, 0.f, 0.f);
}

__global__ __launch_bounds__(256) void f32_to_bf16_kernel(
    const float* __restrict__ in, __bf16* __restrict__ out, long long n4)
{
    long long i = (long long)blockIdx.x * 256 + threadIdx.x;
    if (i >= n4) return;
    float4 v = ((const float4*)in)[i];
    __bf16* o = out + i * 4;
    o[0] = (__bf16)v.x; o[1] = (__bf16)v.y; o[2] = (__bf16)v.z; o[3] = (__bf16)v.w;
}

// segment_sum: agg[dst[e]] += e_feat[e], f32 atomics (agg fits in 192MB L2).
__global__ __launch_bounds__(256) void scatter_add_kernel(
    const float* __restrict__ e_feat, const int* __restrict__ dstI,
    float* __restrict__ agg, int E_)
{
    long long t = (long long)blockIdx.x * 256 + threadIdx.x;
    int edge = (int)(t >> 5);
    int q = (int)(t & 31);
    if (edge >= E_) return;
    int d = dstI[edge];
    float4 v = *(const float4*)(e_feat + (size_t)edge * GN_H + q * 4);
    float* a = agg + (size_t)d * GN_H + q * 4;
    atomicAdd(a + 0, v.x);
    atomicAdd(a + 1, v.y);
    atomicAdd(a + 2, v.z);
    atomicAdd(a + 3, v.w);
}

// ------------------------- fused MLP(+LN+residual) -------------------------
// One workgroup (256 thr = 8 wave32) handles 16 rows x 128 cols.
// Wave w owns output column tile [16*w, 16*w+16). Stage 1: hidden =
// relu(A@W1+b1) accumulated via v_wmma_f32_16x16x32_bf16, staged bf16 in LDS.
// Stage 2: out = hidden@W2+b2, then LayerNorm(+residual) or raw store.
//
// MODE: 0 = encoder (A0 row-major bf16 stride 32, LN, no residual)
//       1 = edge layer (A = [e | v[src] | v[dst]], LN + residual into e)
//       2 = vertex layer (A = [v | agg], LN + residual into v)
//       3 = decoder (A0 stride 128, no LN, 12-col f32 out)
template<int MODE, int NCH>
__global__ __launch_bounds__(256) void fused_mlp_kernel(
    const __bf16* __restrict__ A0, const __bf16* __restrict__ A1,
    const int* __restrict__ srcI, const int* __restrict__ dstI,
    const __bf16* __restrict__ W1p, const float* __restrict__ b1,
    const __bf16* __restrict__ W2p, const float* __restrict__ b2,
    const float* __restrict__ gam, const float* __restrict__ bet,
    const float* __restrict__ resid,
    float* __restrict__ outF, __bf16* __restrict__ outB)
{
    __shared__ alignas(16) __bf16 hid[16 * GN_H];   // 4 KB  (hidden, bf16)
    __shared__ alignas(16) float  obuf[16 * GN_H];  // 8 KB  (pre-LN output)
    __shared__ float psum[16 * 8], psq[16 * 8];
    __shared__ float rmean[16], rrstd[16];

    const int tid  = threadIdx.x;
    const int wave = tid >> 5;        // 0..7  -> output col tile
    const int lane = tid & 31;
    const int m    = lane & 15;       // row within tile (A) / col within tile (B,D)
    const int half = lane >> 4;       // K-half selector (A/B layouts)
    const int row0 = blockIdx.x * 16;
    const int grow = row0 + m;        // this lane's A row
    const int nt   = wave;
    const int ncol = nt * 16 + m;     // this lane's D column

    // ---------------- stage 1: hidden = relu(A @ W1 + b1) ----------------
    f32x8 acc = {};
#pragma unroll
    for (int c = 0; c < NCH; ++c) {
        const __bf16* ap;
        int off;
        if (MODE == 1) {              // [e | v[src] | v[dst]], 12 chunks of 32
            if (c < 4)      { ap = A0 + (size_t)grow * GN_H;        off = c * 32; }
            else if (c < 8) { int s = srcI[grow];
                              ap = A1 + (size_t)s * GN_H;           off = (c - 4) * 32; }
            else            { int d = dstI[grow];
                              ap = A1 + (size_t)d * GN_H;           off = (c - 8) * 32; }
        } else if (MODE == 2) {       // [v | agg], 8 chunks
            if (c < 4)      { ap = A0 + (size_t)grow * GN_H;        off = c * 32; }
            else            { ap = A1 + (size_t)grow * GN_H;        off = (c - 4) * 32; }
        } else if (MODE == 0) {       // padded input, stride 32, 1 chunk
            ap = A0 + (size_t)grow * 32; off = 0;
        } else {                      // decoder: v, stride 128
            ap = A0 + (size_t)grow * GN_H; off = c * 32;
        }
        // A fragment (16x32 bf16): lanes 0-15 K={0..7,16..23}, 16-31 K={8..15,24..31}
        bf16x8 r1 = *(const bf16x8*)(ap + off + half * 8);
        bf16x8 r2 = *(const bf16x8*)(ap + off + half * 8 + 16);
        bf16x16 afrag;
#pragma unroll
        for (int i = 0; i < 8; ++i) { afrag[i] = r1[i]; afrag[8 + i] = r2[i]; }
        // B fragment: pre-packed, one contiguous 32B load per lane
        bf16x16 bfrag = *(const bf16x16*)(W1p + ((size_t)((c * 8 + nt) * 32 + lane) << 4));
        acc = __builtin_amdgcn_wmma_f32_16x16x32_bf16(
            false, afrag, false, bfrag, (short)0, acc, false, false);
    }
    {
        float bias1 = b1[ncol];
#pragma unroll
        for (int r = 0; r < 8; ++r) {
            float h = acc[r] + bias1;
            h = h > 0.0f ? h : 0.0f;
            hid[(r + half * 8) * GN_H + ncol] = (__bf16)h;   // D layout: row r+8*half
        }
    }
    __syncthreads();

    // ---------------- stage 2: out = hidden @ W2 + b2 ----------------
    constexpr int N2T = (MODE == 3) ? 1 : 8;
    const int nt2 = (MODE == 3) ? 0 : nt;
    f32x8 acc2 = {};
#pragma unroll
    for (int c = 0; c < 4; ++c) {
        bf16x8 r1 = *(const bf16x8*)(&hid[m * GN_H + c * 32 + half * 8]);
        bf16x8 r2 = *(const bf16x8*)(&hid[m * GN_H + c * 32 + half * 8 + 16]);
        bf16x16 afrag;
#pragma unroll
        for (int i = 0; i < 8; ++i) { afrag[i] = r1[i]; afrag[8 + i] = r2[i]; }
        bf16x16 bfrag = *(const bf16x16*)(W2p + ((size_t)((c * N2T + nt2) * 32 + lane) << 4));
        acc2 = __builtin_amdgcn_wmma_f32_16x16x32_bf16(
            false, afrag, false, bfrag, (short)0, acc2, false, false);
    }

    if (MODE == 3) {
        // decoder: only col tile 0 holds the 12 real columns
        if (nt == 0 && m < GN_OD) {
            float bias2 = b2[m];
#pragma unroll
            for (int r = 0; r < 8; ++r)
                outF[(size_t)(row0 + r + half * 8) * GN_OD + m] = acc2[r] + bias2;
        }
        return;
    }

    // ---------------- LayerNorm (+ residual) epilogue ----------------
    {
        float bias2 = b2[ncol];
#pragma unroll
        for (int r = 0; r < 8; ++r)
            obuf[(r + half * 8) * GN_H + ncol] = acc2[r] + bias2;
    }
    __syncthreads();
    if (tid < 128) {
        int row = tid >> 3, seg = tid & 7;
        const float* p = &obuf[row * GN_H + seg * 16];
        float s = 0.f, q = 0.f;
#pragma unroll
        for (int i = 0; i < 16; ++i) { float x = p[i]; s += x; q += x * x; }
        psum[row * 8 + seg] = s;
        psq[row * 8 + seg]  = q;
    }
    __syncthreads();
    if (tid < 16) {
        float s = 0.f, q = 0.f;
#pragma unroll
        for (int i = 0; i < 8; ++i) { s += psum[tid * 8 + i]; q += psq[tid * 8 + i]; }
        float mean = s * (1.0f / GN_H);
        float var  = q * (1.0f / GN_H) - mean * mean;
        rmean[tid] = mean;
        rrstd[tid] = rsqrtf(var + 1e-5f);
    }
    __syncthreads();
    {
        int base = tid * 8;
        int row  = base >> 7;       // /128
        int col0 = base & 127;
        size_t gbase = (size_t)(row0 + row) * GN_H + col0;
        float mean = rmean[row], rstd = rrstd[row];
#pragma unroll
        for (int i = 0; i < 8; ++i) {
            int col = col0 + i;
            float x = (obuf[row * GN_H + col] - mean) * rstd * gam[col] + bet[col];
            if (MODE != 0) x += resid[gbase + i];   // residual (in-place safe: own rows)
            outF[gbase + i] = x;
            outB[gbase + i] = (__bf16)x;
        }
    }
}

// ------------------------------- launcher ----------------------------------

extern "C" void kernel_launch(void* const* d_in, const int* in_sizes, int n_in,
                              void* d_out, int out_size, void* d_ws, size_t ws_size,
                              hipStream_t stream)
{
    const float* vertex_attr = (const float*)d_in[0];
    const float* edge_attr   = (const float*)d_in[1];
    const int*   edge_index  = (const int*)d_in[2];
    const int*   srcI = edge_index;
    const int*   dstI = edge_index + GN_E;

    const float* enc_v_W1 = (const float*)d_in[3];
    const float* enc_v_b1 = (const float*)d_in[4];
    const float* enc_v_W2 = (const float*)d_in[5];
    const float* enc_v_b2 = (const float*)d_in[6];
    const float* enc_v_g  = (const float*)d_in[7];
    const float* enc_v_be = (const float*)d_in[8];
    const float* enc_e_W1 = (const float*)d_in[9];
    const float* enc_e_b1 = (const float*)d_in[10];
    const float* enc_e_W2 = (const float*)d_in[11];
    const float* enc_e_b2 = (const float*)d_in[12];
    const float* enc_e_g  = (const float*)d_in[13];
    const float* enc_e_be = (const float*)d_in[14];
    const float* pe_W1 = (const float*)d_in[15];
    const float* pe_b1 = (const float*)d_in[16];
    const float* pe_W2 = (const float*)d_in[17];
    const float* pe_b2 = (const float*)d_in[18];
    const float* pe_g  = (const float*)d_in[19];
    const float* pe_be = (const float*)d_in[20];
    const float* pv_W1 = (const float*)d_in[21];
    const float* pv_b1 = (const float*)d_in[22];
    const float* pv_W2 = (const float*)d_in[23];
    const float* pv_b2 = (const float*)d_in[24];
    const float* pv_g  = (const float*)d_in[25];
    const float* pv_be = (const float*)d_in[26];
    const float* dec_W1 = (const float*)d_in[27];
    const float* dec_b1 = (const float*)d_in[28];
    const float* dec_W2 = (const float*)d_in[29];
    const float* dec_b2 = (const float*)d_in[30];

    float* out = (float*)d_out;

    // -------- carve workspace --------
    char* wsp = (char*)d_ws;
    auto take = [&](size_t bytes) -> void* {
        void* p = (void*)wsp;
        wsp += (bytes + 255) & ~(size_t)255;
        return p;
    };
    float*  vF   = (float*) take((size_t)GN_N * GN_H * 4);
    __bf16* vB   = (__bf16*)take((size_t)GN_N * GN_H * 2);
    float*  eF   = (float*) take((size_t)GN_E * GN_H * 4);
    __bf16* eB   = (__bf16*)take((size_t)GN_E * GN_H * 2);
    float*  aggF = (float*) take((size_t)GN_N * GN_H * 4);
    __bf16* aggB = (__bf16*)take((size_t)GN_N * GN_H * 2);
    __bf16* vaP  = (__bf16*)take((size_t)GN_N * 32 * 2);
    __bf16* eaP  = (__bf16*)take((size_t)GN_E * 32 * 2);
    __bf16* encvW1p = (__bf16*)take((size_t)32 * 128 * 2);
    __bf16* encvW2p = (__bf16*)take((size_t)128 * 128 * 2);
    __bf16* enceW1p = (__bf16*)take((size_t)32 * 128 * 2);
    __bf16* enceW2p = (__bf16*)take((size_t)128 * 128 * 2);
    __bf16* peW1p   = (__bf16*)take((size_t)GN_L * 384 * 128 * 2);
    __bf16* peW2p   = (__bf16*)take((size_t)GN_L * 128 * 128 * 2);
    __bf16* pvW1p   = (__bf16*)take((size_t)GN_L * 256 * 128 * 2);
    __bf16* pvW2p   = (__bf16*)take((size_t)GN_L * 128 * 128 * 2);
    __bf16* decW1p  = (__bf16*)take((size_t)128 * 128 * 2);
    __bf16* decW2p  = (__bf16*)take((size_t)128 * 16 * 2);

    const dim3 blk(256);
    auto cdiv = [](long long a, long long b) { return (unsigned)((a + b - 1) / b); };

    // -------- 0) pack weights into WMMA B-fragment order --------
    pack_b_kernel<<<cdiv(32 * 128, 256), blk, 0, stream>>>(enc_v_W1, encvW1p, GN_VD, 32, 128, 128);
    pack_b_kernel<<<cdiv(128 * 128, 256), blk, 0, stream>>>(enc_v_W2, encvW2p, 128, 128, 128, 128);
    pack_b_kernel<<<cdiv(32 * 128, 256), blk, 0, stream>>>(enc_e_W1, enceW1p, GN_ED, 32, 128, 128);
    pack_b_kernel<<<cdiv(128 * 128, 256), blk, 0, stream>>>(enc_e_W2, enceW2p, 128, 128, 128, 128);
    for (int l = 0; l < GN_L; ++l) {
        pack_b_kernel<<<cdiv(384 * 128, 256), blk, 0, stream>>>(
            pe_W1 + (size_t)l * 384 * 128, peW1p + (size_t)l * 384 * 128, 384, 384, 128, 128);
        pack_b_kernel<<<cdiv(128 * 128, 256), blk, 0, stream>>>(
            pe_W2 + (size_t)l * 128 * 128, peW2p + (size_t)l * 128 * 128, 128, 128, 128, 128);
        pack_b_kernel<<<cdiv(256 * 128, 256), blk, 0, stream>>>(
            pv_W1 + (size_t)l * 256 * 128, pvW1p + (size_t)l * 256 * 128, 256, 256, 128, 128);
        pack_b_kernel<<<cdiv(128 * 128, 256), blk, 0, stream>>>(
            pv_W2 + (size_t)l * 128 * 128, pvW2p + (size_t)l * 128 * 128, 128, 128, 128, 128);
    }
    pack_b_kernel<<<cdiv(128 * 128, 256), blk, 0, stream>>>(dec_W1, decW1p, 128, 128, 128, 128);
    pack_b_kernel<<<cdiv(128 * 16, 256), blk, 0, stream>>>(dec_W2, decW2p, 128, 128, GN_OD, 16);

    // -------- 1) pad+convert raw attributes --------
    pad_rows_kernel<<<cdiv((long long)GN_N * 32, 256), blk, 0, stream>>>(vertex_attr, vaP, GN_N, GN_VD, 32);
    pad_rows_kernel<<<cdiv((long long)GN_E * 32, 256), blk, 0, stream>>>(edge_attr, eaP, GN_E, GN_ED, 32);

    // -------- 2) encoders --------
    fused_mlp_kernel<0, 1><<<GN_N / 16, blk, 0, stream>>>(
        vaP, nullptr, nullptr, nullptr,
        encvW1p, enc_v_b1, encvW2p, enc_v_b2, enc_v_g, enc_v_be,
        nullptr, vF, vB);
    fused_mlp_kernel<0, 1><<<GN_E / 16, blk, 0, stream>>>(
        eaP, nullptr, nullptr, nullptr,
        enceW1p, enc_e_b1, enceW2p, enc_e_b2, enc_e_g, enc_e_be,
        nullptr, eF, eB);

    // -------- 3) processor layers --------
    for (int l = 0; l < GN_L; ++l) {
        const size_t wo1 = (size_t)l * 384 * 128, wo2 = (size_t)l * 128 * 128;
        const size_t vo1 = (size_t)l * 256 * 128;
        const size_t bo = (size_t)l * 128;

        fused_mlp_kernel<1, 12><<<GN_E / 16, blk, 0, stream>>>(
            eB, vB, srcI, dstI,
            peW1p + wo1, pe_b1 + bo, peW2p + wo2, pe_b2 + bo, pe_g + bo, pe_be + bo,
            eF, eF, eB);

        zero_f32_kernel<<<cdiv((long long)GN_N * GN_H / 4, 256), blk, 0, stream>>>(
            aggF, (long long)GN_N * GN_H / 4);
        scatter_add_kernel<<<cdiv((long long)GN_E * 32, 256), blk, 0, stream>>>(
            eF, dstI, aggF, GN_E);
        f32_to_bf16_kernel<<<cdiv((long long)GN_N * GN_H / 4, 256), blk, 0, stream>>>(
            aggF, aggB, (long long)GN_N * GN_H / 4);

        fused_mlp_kernel<2, 8><<<GN_N / 16, blk, 0, stream>>>(
            vB, aggB, nullptr, nullptr,
            pvW1p + vo1, pv_b1 + bo, pvW2p + wo2, pv_b2 + bo, pv_g + bo, pv_be + bo,
            vF, vF, vB);
    }

    // -------- 4) decoder --------
    fused_mlp_kernel<3, 4><<<GN_N / 16, blk, 0, stream>>>(
        vB, nullptr, nullptr, nullptr,
        decW1p, dec_b1, decW2p, dec_b2, nullptr, nullptr,
        nullptr, out, nullptr);
}
